// AttentionPositionalBias_87694642249760
// MI455X (gfx1250) — matrix operand order repaired and checked by reading, not compiled, not run on hardware
//
#include <hip/hip_runtime.h>
#include <hip/hip_bf16.h>
#include <math.h>

// ---- problem constants (from reference) ----
#define BATCH   4
#define SEQ     2048
#define HEADS   8
#define DHEAD   64
#define DMODEL  512              // HEADS * DHEAD
#define ROWS    (BATCH * SEQ)    // 8192
#define QKSCALE 0.125f           // 1/sqrt(64)

// padded LDS strides (halves); byte strides 144 / 80 keep 16B alignment, spread banks
#define KSTR 72
#define VSTR 40
#define PSTR 40

typedef _Float16 v8h  __attribute__((ext_vector_type(8)));
typedef _Float16 v16h __attribute__((ext_vector_type(16)));
typedef float    v8f  __attribute__((ext_vector_type(8)));

// Build a v16h A/B fragment from two contiguous 8-half (16-byte) chunks.
__device__ __forceinline__ v16h mk_frag(const _Float16* p0, const _Float16* p1) {
    v8h lo = *reinterpret_cast<const v8h*>(p0);
    v8h hi = *reinterpret_cast<const v8h*>(p1);
    v16h r;
#pragma unroll
    for (int i = 0; i < 8; ++i) { r[i] = lo[i]; r[i + 8] = hi[i]; }
    return r;
}

__device__ __forceinline__ float rowmax16(float v) {
    v = fmaxf(v, __shfl_xor(v, 1, 16));
    v = fmaxf(v, __shfl_xor(v, 2, 16));
    v = fmaxf(v, __shfl_xor(v, 4, 16));
    v = fmaxf(v, __shfl_xor(v, 8, 16));
    return v;
}
__device__ __forceinline__ float rowsum16(float v) {
    v += __shfl_xor(v, 1, 16);
    v += __shfl_xor(v, 2, 16);
    v += __shfl_xor(v, 4, 16);
    v += __shfl_xor(v, 8, 16);
    return v;
}

#define WMMA_F16(A, B, C) __builtin_amdgcn_wmma_f32_16x16x32_f16( \
        false, (A), false, (B), (short)0, (C), false, false)

// ---------------- kernel 1: f32 -> f16 conversion ----------------
__global__ void cvt_f32_to_f16(const float* __restrict__ src,
                               _Float16* __restrict__ dst, int n) {
    int i = blockIdx.x * blockDim.x + threadIdx.x;
    if (i < n) dst[i] = (_Float16)src[i];
}

// ---------------- kernel 2: build Vt[b][h][d][n] = x[b][n][h*64+d] ----------------
__global__ void build_vt(const float* __restrict__ x, _Float16* __restrict__ vt) {
    int idx = blockIdx.x * blockDim.x + threadIdx.x;      // over BATCH*HEADS*DHEAD*SEQ
    if (idx >= BATCH * HEADS * DHEAD * SEQ) return;
    int j    = idx & (SEQ - 1);
    int rest = idx >> 11;            // /SEQ
    int d    = rest & (DHEAD - 1);
    int bh   = rest >> 6;            // b*HEADS + h
    int b    = bh >> 3, h = bh & 7;
    vt[idx] = (_Float16)x[((size_t)(b * SEQ + j)) * DMODEL + h * DHEAD + d];
}

// ---------------- kernel 3: Q/K projection GEMM via WMMA ----------------
// One wave computes one 16x16 output tile of (x @ W^T), K = 512.
// grid: (ROWS/16/4, DMODEL/16, 2), block: 128 (4 waves)
__global__ __launch_bounds__(128)
void proj_qk(const _Float16* __restrict__ xh,
             const _Float16* __restrict__ wqh,
             const _Float16* __restrict__ wkh,
             _Float16* __restrict__ qh,
             _Float16* __restrict__ kh) {
    const int lane = threadIdx.x & 31;
    const int wave = threadIdx.x >> 5;
    const int col  = lane & 15;      // N index within tile
    const int hi   = lane >> 4;      // lane half

    const int mtile = blockIdx.x * 4 + wave;
    const int gr0   = mtile * 16;                 // first global row (b*SEQ+n flat)
    const int o0    = blockIdx.y * 16;            // first output column
    const _Float16* W   = blockIdx.z ? wkh : wqh;
    _Float16*       Out = blockIdx.z ? kh  : qh;

    v8f acc = {};
    const _Float16* arow = xh + (size_t)(gr0 + col) * DMODEL;
    const _Float16* brow = W  + (size_t)(o0  + col) * DMODEL;

#pragma unroll 4
    for (int kk = 0; kk < DMODEL; kk += 32) {
        // A fragment: row m = col, k runs [kk+hi*8, +8) and [kk+16+hi*8, +8)
        v16h af = mk_frag(arow + kk + hi * 8, arow + kk + 16 + hi * 8);
        // B fragment: col n = col, k runs [kk+hi*8, +8) and [kk+16+hi*8, +8)
        v16h bf = mk_frag(brow + kk + hi * 8, brow + kk + 16 + hi * 8);
        acc = WMMA_F16(af, bf, acc);
    }

    // Scatter C tile (row M = r + 8*hi, col = o0+col) into head-major [b][h][n][64] f16
    const int o = o0 + col;
    const int h = o >> 6, d = o & 63;
#pragma unroll
    for (int r = 0; r < 8; ++r) {
        int gr = gr0 + r + 8 * hi;
        int b  = gr >> 11;           // /SEQ
        int i  = gr & (SEQ - 1);
        Out[((size_t)(b * HEADS + h) * SEQ + i) * DHEAD + d] = (_Float16)acc[r];
    }
}

// ---------------- kernel 4: flash attention with ALiBi ----------------
// Block = 4 waves, all on the same (b,h); each wave owns 16 query rows.
// K/V tiles (shared by all 4 waves) are double-buffer staged into LDS with
// GLOBAL_LOAD_ASYNC_TO_LDS_B128 (ASYNCcnt) overlapping the WMMA work.
// grid: (SEQ/16/4, BATCH*HEADS), block: 128 (4 waves)
__global__ __launch_bounds__(128)
void attn_flash(const _Float16* __restrict__ qh,
                const _Float16* __restrict__ kh,
                const _Float16* __restrict__ vt,
                const float* __restrict__ positions,
                float* __restrict__ out) {
    __shared__ _Float16 Klds[2][32 * KSTR];   // K tile: rows j_local(32) x d(64), padded
    __shared__ _Float16 Vlds[2][64 * VSTR];   // V tile: rows d(64) x j_local(32), padded
    __shared__ _Float16 Plds[4][16 * PSTR];   // per-wave P staging (C->A relayout)

    const int tid  = threadIdx.x;
    const int lane = tid & 31;
    const int wave = tid >> 5;
    const int col  = lane & 15;
    const int hi   = lane >> 4;

    const int i0 = (blockIdx.x * 4 + wave) * 16;
    const int bh = blockIdx.y;
    const int b  = bh >> 3, h = bh & 7;
    const float slope = exp2f(-(float)(h + 1));  // ALiBi slopes for 8 heads

    const _Float16* kbh = kh + (size_t)bh * SEQ * DHEAD;
    const _Float16* vbh = vt + (size_t)bh * DHEAD * SEQ;

    // Stage one 32-key tile (K: 32x64 f16, V: 64x32 f16) into LDS buffer `buf`.
    // 512 x 16B chunks over 128 threads = 4 async b128 ops per thread.
    auto stage = [&](int j0, int buf) {
#pragma unroll
        for (int c = 0; c < 2; ++c) {
            int q = tid * 2 + c;                                 // 0..255
            const _Float16* src = kbh + (size_t)(j0 + (q >> 3)) * DHEAD + (q & 7) * 8;
            unsigned dst = (unsigned)(uintptr_t)&Klds[buf][(q >> 3) * KSTR + (q & 7) * 8];
            asm volatile("global_load_async_to_lds_b128 %0, %1, off"
                         :: "v"(dst), "v"((unsigned long long)(uintptr_t)src)
                         : "memory");
        }
#pragma unroll
        for (int c = 0; c < 2; ++c) {
            int q = tid * 2 + c;                                 // 0..255
            const _Float16* src = vbh + (size_t)(q >> 2) * SEQ + j0 + (q & 3) * 8;
            unsigned dst = (unsigned)(uintptr_t)&Vlds[buf][(q >> 2) * VSTR + (q & 3) * 8];
            asm volatile("global_load_async_to_lds_b128 %0, %1, off"
                         :: "v"(dst), "v"((unsigned long long)(uintptr_t)src)
                         : "memory");
        }
    };

    // --- load Q fragments for this 16-row tile (kept in registers) ---
    const _Float16* qbase = qh + ((size_t)bh * SEQ + i0) * DHEAD + (size_t)col * DHEAD;
    v16h qf0 = mk_frag(qbase + 0  + hi * 16, qbase + 8  + hi * 16);   // d = 0..31
    v16h qf1 = mk_frag(qbase + 32 + hi * 16, qbase + 40 + hi * 16);   // d = 32..63

    // query positions: 8 rows per lane (row M = r + 8*hi), replicated over 16 lanes
    float pi[8];
#pragma unroll
    for (int r = 0; r < 8; ++r) pi[r] = positions[b * SEQ + i0 + r + 8 * hi];

    float mrow[8], lrow[8];
#pragma unroll
    for (int r = 0; r < 8; ++r) { mrow[r] = -1e30f; lrow[r] = 0.0f; }
    v8f o0v = {}, o1v = {}, o2v = {}, o3v = {};

    _Float16* pw = &Plds[wave][0];

    // prologue: stage tile 0, wait, sync
    stage(0, 0);
    asm volatile("s_wait_asynccnt 0x0" ::: "memory");
    __syncthreads();

    int p = 0;
    for (int j0 = 0; j0 < SEQ; j0 += 32) {
        // overlap: kick off async DMA of the next tile into the other buffer
        if (j0 + 32 < SEQ) stage(j0 + 32, p ^ 1);

        const _Float16* kc = &Klds[p][0];
        const _Float16* vc = &Vlds[p][0];

        // ---- S = Q K^T for two 16-col tiles (from LDS) ----
        v8f s0 = {}, s1 = {};
        {
            const _Float16* k0 = kc + (size_t)col * KSTR;          // cols 0..15
            const _Float16* k1 = kc + (size_t)(16 + col) * KSTR;   // cols 16..31
            v16h kb;
            kb = mk_frag(k0 + 0  + hi * 16, k0 + 8  + hi * 16);
            s0 = WMMA_F16(qf0, kb, s0);
            kb = mk_frag(k0 + 32 + hi * 16, k0 + 40 + hi * 16);
            s0 = WMMA_F16(qf1, kb, s0);
            kb = mk_frag(k1 + 0  + hi * 16, k1 + 8  + hi * 16);
            s1 = WMMA_F16(qf0, kb, s1);
            kb = mk_frag(k1 + 32 + hi * 16, k1 + 40 + hi * 16);
            s1 = WMMA_F16(qf1, kb, s1);
        }

        // ---- online softmax with ALiBi bias, write P (f16) to LDS ----
        const float pj0 = positions[b * SEQ + j0 + col];
        const float pj1 = positions[b * SEQ + j0 + 16 + col];
#pragma unroll
        for (int r = 0; r < 8; ++r) {
            float v0 = s0[r] * QKSCALE - slope * fabsf(pj0 - pi[r]);
            float v1 = s1[r] * QKSCALE - slope * fabsf(pj1 - pi[r]);
            float rm = rowmax16(fmaxf(v0, v1));
            float nm = fmaxf(mrow[r], rm);
            float sf = __expf(mrow[r] - nm);
            float p0 = __expf(v0 - nm);
            float p1 = __expf(v1 - nm);
            lrow[r] = lrow[r] * sf + rowsum16(p0 + p1);
            mrow[r] = nm;
            o0v[r] *= sf; o1v[r] *= sf; o2v[r] *= sf; o3v[r] *= sf;
            int row = r + 8 * hi;
            pw[row * PSTR + col]      = (_Float16)p0;
            pw[row * PSTR + 16 + col] = (_Float16)p1;
        }

        // wave-local LDS RAW fence before re-reading P in A-fragment layout
        asm volatile("s_wait_dscnt 0x0" ::: "memory");

        // P as A fragment: row m = col, k run [hi*16,+16)
        v16h pa = mk_frag(pw + col * PSTR + hi * 16, pw + col * PSTR + hi * 16 + 8);

        // ---- O += P @ V (4 d-chunks of 16, from LDS) ----
        {
            v16h vf;
            const _Float16* vb;
            vb = vc + (size_t)(0 * 16 + col) * VSTR + hi * 16;
            vf = mk_frag(vb, vb + 8);
            o0v = WMMA_F16(pa, vf, o0v);
            vb = vc + (size_t)(1 * 16 + col) * VSTR + hi * 16;
            vf = mk_frag(vb, vb + 8);
            o1v = WMMA_F16(pa, vf, o1v);
            vb = vc + (size_t)(2 * 16 + col) * VSTR + hi * 16;
            vf = mk_frag(vb, vb + 8);
            o2v = WMMA_F16(pa, vf, o2v);
            vb = vc + (size_t)(3 * 16 + col) * VSTR + hi * 16;
            vf = mk_frag(vb, vb + 8);
            o3v = WMMA_F16(pa, vf, o3v);
        }

        // ensure this wave's async stage of tile j+1 landed, then block-wide sync:
        // everyone done reading buffer p AND buffer p^1 fully staged.
        asm volatile("s_wait_asynccnt 0x0" ::: "memory");
        __syncthreads();
        p ^= 1;
    }

    // ---- normalize and store out[b][i][h*64+d] (f32) ----
#pragma unroll
    for (int r = 0; r < 8; ++r) {
        float inv = 1.0f / lrow[r];
        int i = i0 + r + 8 * hi;
        float* ob = out + ((size_t)(b * SEQ + i)) * DMODEL + h * DHEAD + col;
        ob[0]  = o0v[r] * inv;
        ob[16] = o1v[r] * inv;
        ob[32] = o2v[r] * inv;
        ob[48] = o3v[r] * inv;
    }
}

// ---------------- host-side launch ----------------
extern "C" void kernel_launch(void* const* d_in, const int* in_sizes, int n_in,
                              void* d_out, int out_size, void* d_ws, size_t ws_size,
                              hipStream_t stream) {
    const float* x   = (const float*)d_in[0];   // [4,2048,512]
    const float* pos = (const float*)d_in[1];   // [4,2048]
    const float* Wq  = (const float*)d_in[2];   // [512,512]
    const float* Wk  = (const float*)d_in[3];   // [512,512]
    float* out = (float*)d_out;

    // workspace layout (all f16), ~34.6 MB total
    char* ws = (char*)d_ws;
    size_t off = 0;
    _Float16* xh  = (_Float16*)(ws + off); off += (size_t)ROWS * DMODEL * 2;         // 8.4 MB
    _Float16* wqh = (_Float16*)(ws + off); off += (size_t)DMODEL * DMODEL * 2;       // 0.5 MB
    _Float16* wkh = (_Float16*)(ws + off); off += (size_t)DMODEL * DMODEL * 2;       // 0.5 MB
    _Float16* qh  = (_Float16*)(ws + off); off += (size_t)ROWS * DMODEL * 2;         // 8.4 MB
    _Float16* kh  = (_Float16*)(ws + off); off += (size_t)ROWS * DMODEL * 2;         // 8.4 MB
    _Float16* vt  = (_Float16*)(ws + off); off += (size_t)ROWS * DMODEL * 2;         // 8.4 MB
    (void)ws_size; (void)in_sizes; (void)n_in; (void)out_size;

    const int nx = ROWS * DMODEL;      // 4,194,304
    const int nw = DMODEL * DMODEL;    // 262,144

    cvt_f32_to_f16<<<dim3((nx + 255) / 256), dim3(256), 0, stream>>>(x,  xh,  nx);
    cvt_f32_to_f16<<<dim3((nw + 255) / 256), dim3(256), 0, stream>>>(Wq, wqh, nw);
    cvt_f32_to_f16<<<dim3((nw + 255) / 256), dim3(256), 0, stream>>>(Wk, wkh, nw);
    build_vt<<<dim3((nx + 255) / 256), dim3(256), 0, stream>>>(x, vt);

    // projection: (ROWS/16)/4 = 128 blocks in x, 32 col-tiles, z selects Q/K
    proj_qk<<<dim3(ROWS / 16 / 4, DMODEL / 16, 2), dim3(128), 0, stream>>>(
        xh, wqh, wkh, qh, kh);

    // attention: (SEQ/16)/4 = 32 blocks in x, BATCH*HEADS = 32 in y
    attn_flash<<<dim3(SEQ / 16 / 4, BATCH * HEADS), dim3(128), 0, stream>>>(
        qh, kh, vt, pos, out);
}
